// ImplicitResampleModule_31696858644760
// MI455X (gfx1250) — compile-verified, matrix operand order checked by reading, not run on hardware
//
#include <hip/hip_runtime.h>
#include <hip/hip_bf16.h>
#include <math.h>

typedef __attribute__((ext_vector_type(16))) _Float16 v16h;
typedef __attribute__((ext_vector_type(8)))  _Float16 v8h;
typedef __attribute__((ext_vector_type(8)))  float    v8f;

#define BB    2
#define CC    256
#define HH    256
#define WW    256
#define THH   256
#define TWW   256
#define UVN   4
#define PED   256
#define DIMX  256
#define NHEAD 8
#define DH    32
#define PIX   16            // pixels per workgroup
#define LDA   264           // 256 + 8 halves pad (bank-conflict avoidance)
#define WSZ   (PED * DIMX)  // 65536 elements per weight matrix

#if __has_builtin(__builtin_amdgcn_wmma_f16_16x16x32_f16)
#define HAVE_WMMA_F16C 1
#else
#define HAVE_WMMA_F16C 0
#endif

// ---------------------------------------------------------------------------
// Pre-pass: convert the three f32 weight matrices to f16 once into d_ws.
// Requires ws_size >= 3*WSZ*2 = 384 KB.
// ---------------------------------------------------------------------------
extern "C" __global__ __launch_bounds__(256)
void wcvt(const float* __restrict__ Wq, const float* __restrict__ Wk,
          const float* __restrict__ Wv, _Float16* __restrict__ w16)
{
    const int i = (blockIdx.x * 256 + threadIdx.x) * 4;
#pragma unroll
    for (int j = 0; j < 4; ++j) {
        w16[i + j]           = (_Float16)Wq[i + j];
        w16[WSZ + i + j]     = (_Float16)Wk[i + j];
        w16[2 * WSZ + i + j] = (_Float16)Wv[i + j];
    }
}

// A-fragment (16x32 f16) for v_wmma_*_16x16x32_f16 from LDS.
// lane<16:  row=lane,    K = {kb..kb+7, kb+16..kb+23},  kb = ks*32
// lane>=16: row=lane-16, K = {kb+8..kb+15, kb+24..kb+31}
__device__ __forceinline__ v16h lds_afrag(const _Float16* base, int rowBase, int ks, int lane) {
    const _Float16* rp = base + (rowBase + (lane & 15)) * LDA
                              + ks * 32 + ((lane & 16) ? 8 : 0);
    v8h lo = *(const v8h*)rp;
    v8h hi = *(const v8h*)(rp + 16);
    return __builtin_shufflevector(lo, hi, 0,1,2,3,4,5,6,7,8,9,10,11,12,13,14,15);
}

// B-fragment (32x16 K x N) from the pre-converted f16 weights.
// Column n of B is row n of W (out = A @ W^T): two contiguous 16-B loads.
__device__ __forceinline__ v16h glb_bfrag16(const _Float16* __restrict__ Wn, int ks, int lane) {
    const _Float16* p = Wn + ks * 32 + ((lane & 16) ? 8 : 0);
    v8h lo = *(const v8h*)p;
    v8h hi = *(const v8h*)(p + 16);
    return __builtin_shufflevector(lo, hi, 0,1,2,3,4,5,6,7,8,9,10,11,12,13,14,15);
}

extern "C" __global__ __launch_bounds__(256, 1)
void irm_fused(const float* __restrict__ feat,
               const float* __restrict__ offset,
               const _Float16* __restrict__ w16,
               const float* __restrict__ bq,
               const float* __restrict__ bk,
               const float* __restrict__ bv,
               float* __restrict__ out)
{
    __shared__ __align__(16) _Float16 sA_fw[64 * LDA];   // key/value input rows
    __shared__ __align__(16) _Float16 sA_fb[16 * LDA];   // query input rows
    __shared__ __align__(16) _Float16 sQ[16 * LDA];
    __shared__ __align__(16) _Float16 sK[64 * LDA];
    __shared__ __align__(16) _Float16 sV[64 * LDA];
    __shared__ __align__(16) _Float16 sPB[UVN * 256];    // window sine PE (block-constant)
    __shared__ __align__(16) float    sInvT[256];        // 1/dim_t per channel
    __shared__ __align__(16) float    sAW[PIX * NHEAD * UVN]; // softmax weights

    const int tid  = threadIdx.x;
    const int lane = tid & 31;
    const int wv   = tid >> 5;           // wave 0..7
    const int b    = blockIdx.x >> 12;   // 4096 workgroups per batch
    const int pix0 = (blockIdx.x & 4095) * PIX;

    const _Float16* Wq16 = w16;
    const _Float16* Wk16 = w16 + WSZ;
    const _Float16* Wv16 = w16 + 2 * WSZ;

    // Warm the (L2-resident) f16 weights via the gfx1250 prefetch path.
    __builtin_prefetch(Wq16 + (size_t)tid * 256, 0, 0);
    __builtin_prefetch(Wk16 + (size_t)tid * 256, 0, 0);
    __builtin_prefetch(Wv16 + (size_t)tid * 256, 0, 0);

    // ---------- Phase 0: per-block constant tables (1 channel per thread) ----------
    {
        const float TWO_PI = 6.2831853071795864f;
        const float winN   = TWO_PI / (1.0f + 1e-6f);   // window PE: card = U-1 = 1
        const int   c      = tid;
        const float invt   = __expf(-0.14391156f * (float)((c & 127) >> 1));
        sInvT[c] = invt;
#pragma unroll
        for (int u = 0; u < UVN; ++u) {
            const int uy = u >> 1, ux = u & 1;
            const float base = (c < 128) ? winN * (float)uy : winN * (float)ux;
            const float arg  = base * invt;
            sPB[u * 256 + c] = (_Float16)((c & 1) ? __cosf(arg) : __sinf(arg));
        }
    }
    __syncthreads();

    // ---------- Phase A: gather + sine PE -> LDS A matrices (b128 stores) ----------
    {
        const int p  = tid >> 4;           // pixel within tile
        const int c0 = (tid & 15) * 16;    // 16 contiguous channels per thread
        const int t  = pix0 + p;
        const int th = t >> 8;
        const int tw = t & 255;

        // full = tgt + offset[..., ::-1]
        const float offx = offset[(((size_t)b * THH + th) * TWW + tw) * 2 + 0];
        const float offy = offset[(((size_t)b * THH + th) * TWW + tw) * 2 + 1];
        const float fy = (float)th + offy;
        const float fx = (float)tw + offx;
        const float fiy = floorf(fy), fix = floorf(fx);
        const int   iy = (int)fiy,    ix = (int)fix;

        const float TWO_PI = 6.2831853071795864f;
        const float ynC = (fy - fiy) * (TWO_PI / (2.0f + 1e-6f));
        const float xnC = (fx - fix) * (TWO_PI / (2.0f + 1e-6f));

        int hcl[2], wcl[2];
        hcl[0] = min(max(iy,     0), HH - 1);
        hcl[1] = min(max(iy + 1, 0), HH - 1);
        wcl[0] = min(max(ix,     0), WW - 1);
        wcl[1] = min(max(ix + 1, 0), WW - 1);

        v8h pkFW[UVN][2];   // packed halves per window row
        v8h pkFB[2];        // packed halves for the query row
#pragma unroll
        for (int cc = 0; cc < 16; ++cc) {
            const int c = c0 + cc;
            const float* fc = feat + ((size_t)b * CC + c) * (size_t)(HH * WW);
            float bsum = 0.f;
#pragma unroll
            for (int u = 0; u < UVN; ++u) {
                const int uy = u >> 1, ux = u & 1;
                const float f = fc[hcl[uy] * WW + wcl[ux]];
                bsum += f;
                pkFW[u][cc >> 3][cc & 7] = (_Float16)(f + (float)sPB[u * 256 + c]);
            }
            const float base = (c < 128) ? ynC : xnC;
            const float arg  = base * sInvT[c];
            const float pec  = (c & 1) ? __cosf(arg) : __sinf(arg);
            pkFB[cc >> 3][cc & 7] = (_Float16)(bsum * 0.25f + pec);
        }
#pragma unroll
        for (int u = 0; u < UVN; ++u) {
            *(v8h*)&sA_fw[(p * UVN + u) * LDA + c0]     = pkFW[u][0];
            *(v8h*)&sA_fw[(p * UVN + u) * LDA + c0 + 8] = pkFW[u][1];
        }
        *(v8h*)&sA_fb[p * LDA + c0]     = pkFB[0];
        *(v8h*)&sA_fb[p * LDA + c0 + 8] = pkFB[1];
    }
    __syncthreads();

    // ---------- Phase B: fused q/k/v projections via v_wmma ----------
    // Q accumulates in f32 (feeds softmax logits); K/V accumulate in f16 C/D,
    // halving accumulator VGPRs (no >256-VGPR MSB mode switches) and making the
    // store epilogue conversion-free. 9 independent accumulators per k-step ->
    // no WMMA->WMMA RAW hazard NOPs; each fw A-fragment feeds both K and V.
#pragma unroll 1
    for (int tt = 0; tt < 2; ++tt) {
        const int nt   = wv + tt * 8;             // N-tile 0..15
        const int n    = nt * 16 + (lane & 15);   // output channel of this lane/column
        const int rOff = (lane & 16) ? 8 : 0;     // C/D row offset (both layouts)
        const int colL = lane & 15;
        const _Float16* WqN = Wq16 + (size_t)n * PED;
        const _Float16* WkN = Wk16 + (size_t)n * PED;
        const _Float16* WvN = Wv16 + (size_t)n * PED;

        v8f accQ;
        {
            const float bqv = bq[n];
#pragma unroll
            for (int r = 0; r < 8; ++r) accQ[r] = bqv;
        }

#if HAVE_WMMA_F16C
        v8h accK[4], accV[4];
        {
            const _Float16 bkv = (_Float16)bk[n], bvv = (_Float16)bv[n];
#pragma unroll
            for (int m = 0; m < 4; ++m)
#pragma unroll
                for (int e = 0; e < 8; ++e) { accK[m][e] = bkv; accV[m][e] = bvv; }
        }
#else
        v8f accK[4], accV[4];
        {
            const float bkv = bk[n], bvv = bv[n];
#pragma unroll
            for (int m = 0; m < 4; ++m)
#pragma unroll
                for (int r = 0; r < 8; ++r) { accK[m][r] = bkv; accV[m][r] = bvv; }
        }
#endif

        for (int ks = 0; ks < 8; ++ks) {
            const v16h bmQ = glb_bfrag16(WqN, ks, lane);
            const v16h bmK = glb_bfrag16(WkN, ks, lane);
            const v16h bmV = glb_bfrag16(WvN, ks, lane);
            const v16h aQ  = lds_afrag(sA_fb, 0, ks, lane);
            accQ = __builtin_amdgcn_wmma_f32_16x16x32_f16(false, aQ, false, bmQ,
                                                          (short)0, accQ, false, false);
#pragma unroll
            for (int m = 0; m < 4; ++m) {
                const v16h a = lds_afrag(sA_fw, m * 16, ks, lane);
#if HAVE_WMMA_F16C
                accK[m] = __builtin_amdgcn_wmma_f16_16x16x32_f16(false, a, false, bmK,
                                                                 (short)0, accK[m], false, false);
                accV[m] = __builtin_amdgcn_wmma_f16_16x16x32_f16(false, a, false, bmV,
                                                                 (short)0, accV[m], false, false);
#else
                accK[m] = __builtin_amdgcn_wmma_f32_16x16x32_f16(false, a, false, bmK,
                                                                 (short)0, accK[m], false, false);
                accV[m] = __builtin_amdgcn_wmma_f32_16x16x32_f16(false, a, false, bmV,
                                                                 (short)0, accV[m], false, false);
#endif
            }
        }

#pragma unroll
        for (int r = 0; r < 8; ++r)
            sQ[(r + rOff) * LDA + nt * 16 + colL] = (_Float16)accQ[r];

#if HAVE_WMMA_F16C
        // 16-bit C/D layout: lane<16 holds rows 0..7 (elements 0..7),
        // lane>=16 holds rows 8..15 -> row = m*16 + rOff + e, direct f16 store.
#pragma unroll
        for (int m = 0; m < 4; ++m)
#pragma unroll
            for (int e = 0; e < 8; ++e) {
                sK[(m * 16 + rOff + e) * LDA + nt * 16 + colL] = accK[m][e];
                sV[(m * 16 + rOff + e) * LDA + nt * 16 + colL] = accV[m][e];
            }
#else
#pragma unroll
        for (int m = 0; m < 4; ++m)
#pragma unroll
            for (int r = 0; r < 8; ++r) {
                sK[(m * 16 + r + rOff) * LDA + nt * 16 + colL] = (_Float16)accK[m][r];
                sV[(m * 16 + r + rOff) * LDA + nt * 16 + colL] = (_Float16)accV[m][r];
            }
#endif
    }
    __syncthreads();

    // ---------- Phase C1: softmax weights per (pixel, head), wide LDS loads -------
    if (tid < PIX * NHEAD) {
        const int p = tid >> 3;
        const int h = tid & 7;
        const float scale = 0.17677669529663687f;   // 32^-0.5

        const _Float16* qp = &sQ[p * LDA + h * DH];
        v8h qv[4];
#pragma unroll
        for (int j = 0; j < 4; ++j) qv[j] = *(const v8h*)(qp + j * 8);

        float s[4];
#pragma unroll
        for (int u = 0; u < 4; ++u) {
            const _Float16* kp = &sK[(p * 4 + u) * LDA + h * DH];
            float dot = 0.f;
#pragma unroll
            for (int j = 0; j < 4; ++j) {
                const v8h kv = *(const v8h*)(kp + j * 8);
#pragma unroll
                for (int e = 0; e < 8; ++e)
                    dot += (float)qv[j][e] * (float)kv[e];
            }
            s[u] = dot * scale;
        }
        const float mx = fmaxf(fmaxf(s[0], s[1]), fmaxf(s[2], s[3]));
        float e0 = __expf(s[0] - mx), e1 = __expf(s[1] - mx);
        float e2 = __expf(s[2] - mx), e3 = __expf(s[3] - mx);
        const float inv = 1.f / (e0 + e1 + e2 + e3);
        sAW[(p * NHEAD + h) * 4 + 0] = e0 * inv;
        sAW[(p * NHEAD + h) * 4 + 1] = e1 * inv;
        sAW[(p * NHEAD + h) * 4 + 2] = e2 * inv;
        sAW[(p * NHEAD + h) * 4 + 3] = e3 * inv;
    }
    __syncthreads();

    // ---------- Phase C2: weighted-V output, coalesced channel-major store ----------
    // 16-lane groups sweep 16 consecutive tw for a fixed channel -> 64-B segments.
    {
        const int p  = tid & 15;
        const int t  = pix0 + p;
        const int th = t >> 8, tw = t & 255;
#pragma unroll
        for (int it = 0; it < 16; ++it) {
            const int ch = (tid >> 4) + it * 16;
            const int h  = ch >> 5;   // DH = 32
            const float* aw = &sAW[(p * NHEAD + h) * 4];
            float o = 0.f;
#pragma unroll
            for (int u = 0; u < 4; ++u)
                o += aw[u] * (float)sV[(p * 4 + u) * LDA + ch];
            out[(((size_t)b * DIMX + ch) * THH + th) * TWW + tw] = o;
        }
    }
}

extern "C" void kernel_launch(void* const* d_in, const int* in_sizes, int n_in,
                              void* d_out, int out_size, void* d_ws, size_t ws_size,
                              hipStream_t stream) {
    const float* feat   = (const float*)d_in[0];
    const float* offset = (const float*)d_in[1];
    const float* Wq     = (const float*)d_in[2];
    const float* bq     = (const float*)d_in[3];
    const float* Wk     = (const float*)d_in[4];
    const float* bk     = (const float*)d_in[5];
    const float* Wv     = (const float*)d_in[6];
    const float* bv     = (const float*)d_in[7];
    float*       out    = (float*)d_out;
    _Float16*    w16    = (_Float16*)d_ws;     // needs 384 KB of workspace

    // 1) fold the f32->f16 weight conversion out of the hot loop
    hipLaunchKernelGGL(wcvt, dim3(WSZ / (256 * 4)), dim3(256), 0, stream,
                       Wq, Wk, Wv, w16);

    // 2) fused gather + PE + q/k/v WMMA GEMMs + attention
    dim3 grid(BB * (THH * TWW / PIX));   // 8192 workgroups, 16 pixels each
    dim3 block(256);                     // 8 wave32s
    hipLaunchKernelGGL(irm_fused, grid, block, 0, stream,
                       feat, offset, w16, bq, bk, bv, out);
}
